// ts_Learner_70351564309180
// MI455X (gfx1250) — matrix operand order repaired and checked by reading, not compiled, not run on hardware
//
#include <hip/hip_runtime.h>

// ---------------------------------------------------------------------------
// Types for CDNA5 WMMA (wave32): v_wmma_f32_16x16x32_bf16
// ---------------------------------------------------------------------------
typedef __attribute__((ext_vector_type(16))) __bf16          v16bf;
typedef __attribute__((ext_vector_type(8)))  float           v8f;
typedef __attribute__((ext_vector_type(8)))  unsigned short  v8us;

#define B_DIM   128
#define S_DIM   1024
#define FEAT    256
#define HID     512
#define XDIM    512      // 2*FEAT
#define LSTR    264      // padded LDS row stride (elements)
#define NBLK    64       // persistent LSTM blocks (4 waves each -> 256 waves)

__device__ __forceinline__ unsigned short f32_to_bf16(float f) {
  unsigned u = __float_as_uint(f);
  u += 0x7FFFu + ((u >> 16) & 1u);          // round-to-nearest-even
  return (unsigned short)(u >> 16);
}

__device__ __forceinline__ v8f wmma_bf16(v8f c, v16bf a, v16bf b) {
  return __builtin_amdgcn_wmma_f32_16x16x32_bf16(
      /*neg_a=*/false, a, /*neg_b=*/false, b,
      /*c_mod=*/(short)0, c, /*reuse_a=*/false, /*reuse_b=*/false);
}

// A fragment (16x32 bf16 tile) from row-major bf16 activations [rows, ldk].
// ISA layout: lane l row M=l&15; K = (l>=16?8:0) + (e&7) + (e>=8?16:0).
__device__ __forceinline__ v16bf load_a_frag(const unsigned short* x, int ldk,
                                             int m0, int k0, int lane) {
  const unsigned short* p =
      x + (size_t)(m0 + (lane & 15)) * ldk + k0 + ((lane >> 4) << 3);
  union { v8us h[2]; v16bf b; } u;
  u.h[0] = *reinterpret_cast<const v8us*>(p);
  u.h[1] = *reinterpret_cast<const v8us*>(p + 16);
  return u.b;
}

// B fragment (32x16 bf16 tile) from row-major bf16 weights [out, ldk].
// ISA layout: lane l col N=l&15; K = (l>=16?8:0) + (e&7) + (e>=8?16:0).
__device__ __forceinline__ v16bf load_b_frag(const unsigned short* w, int ldk,
                                             int n0, int k0, int lane) {
  const unsigned short* p =
      w + (size_t)(n0 + (lane & 15)) * ldk + k0 + ((lane >> 4) << 3);
  union { v8us h[2]; v16bf b; } u;
  u.h[0] = *reinterpret_cast<const v8us*>(p);
  u.h[1] = *reinterpret_cast<const v8us*>(p + 16);
  return u.b;
}

// ---------------------------------------------------------------------------
// ReLU + LayerNorm over 16 C-tiles (16 tokens x 256 features) held in-wave.
// Row M of C lives in lanes [0..15] (M=r) / [16..31] (M=8+r): reduce across
// the 16-lane half-group with shfl_xor.
// ---------------------------------------------------------------------------
__device__ __forceinline__ void relu_ln(v8f acc[16], const float* bias, int lane) {
  #pragma unroll
  for (int nt = 0; nt < 16; ++nt) {
    float bn = bias[nt * 16 + (lane & 15)];
    #pragma unroll
    for (int r = 0; r < 8; ++r) {
      float v = acc[nt][r] + bn;
      acc[nt][r] = v > 0.f ? v : 0.f;
    }
  }
  #pragma unroll
  for (int r = 0; r < 8; ++r) {
    float s = 0.f, q = 0.f;
    #pragma unroll
    for (int nt = 0; nt < 16; ++nt) { float v = acc[nt][r]; s += v; q += v * v; }
    #pragma unroll
    for (int m = 1; m < 16; m <<= 1) {
      s += __shfl_xor(s, m, 16);
      q += __shfl_xor(q, m, 16);
    }
    float mu  = s * (1.f / 256.f);
    float var = q * (1.f / 256.f) - mu * mu;
    float rs  = rsqrtf(var + 1e-5f);
    #pragma unroll
    for (int nt = 0; nt < 16; ++nt) acc[nt][r] = (acc[nt][r] - mu) * rs;
  }
}

// One branch: [16 tok x 64] -> 256 (relu,LN) -> 256 (relu,LN), write bf16
// into Xbf[s][b][colbase..colbase+255].
__device__ __forceinline__ void mlp_branch(const v16bf a0[2],
                                           const unsigned short* w0, const float* b0,
                                           const unsigned short* w1, const float* b1,
                                           unsigned short* lds,         // 16*LSTR
                                           unsigned short* xbf, int tok0,
                                           int colbase, int lane) {
  v8f acc[16];
  // layer 0: K=64 (2 chunks)
  #pragma unroll
  for (int nt = 0; nt < 16; ++nt) {
    v16bf b0f = load_b_frag(w0, 64, nt * 16, 0, lane);
    v16bf b1f = load_b_frag(w0, 64, nt * 16, 32, lane);
    v8f c = {};
    c = wmma_bf16(c, a0[0], b0f);
    c = wmma_bf16(c, a0[1], b1f);
    acc[nt] = c;
  }
  relu_ln(acc, b0, lane);
  // stash normalized bf16 to LDS (C-layout -> row-major for next A frags)
  #pragma unroll
  for (int nt = 0; nt < 16; ++nt)
    #pragma unroll
    for (int r = 0; r < 8; ++r) {
      int M = ((lane >> 4) << 3) + r, N = nt * 16 + (lane & 15);
      lds[M * LSTR + N] = f32_to_bf16(acc[nt][r]);
    }
  // CDNA5 split dep counter: make LDS stores visible before ds_load_b128s
  asm volatile("s_wait_dscnt 0" ::: "memory");
  // layer 1: K=256 (8 chunks)
  v16bf a1[8];
  #pragma unroll
  for (int kc = 0; kc < 8; ++kc) a1[kc] = load_a_frag(lds, LSTR, 0, kc * 32, lane);
  #pragma unroll
  for (int nt = 0; nt < 16; ++nt) {
    v8f c = {};
    #pragma unroll
    for (int kc = 0; kc < 8; ++kc)
      c = wmma_bf16(c, a1[kc], load_b_frag(w1, 256, nt * 16, kc * 32, lane));
    acc[nt] = c;
  }
  relu_ln(acc, b1, lane);
  // write to Xbf[(s*B + b)*512 + colbase + N]
  #pragma unroll
  for (int nt = 0; nt < 16; ++nt)
    #pragma unroll
    for (int r = 0; r < 8; ++r) {
      int M = ((lane >> 4) << 3) + r, N = nt * 16 + (lane & 15);
      int t = tok0 + M;
      int bb = t >> 10, ss = t & (S_DIM - 1);      // tok = b*S + s
      xbf[((size_t)ss * B_DIM + bb) * XDIM + colbase + N] = f32_to_bf16(acc[nt][r]);
    }
}

// ---------------------------------------------------------------------------
// Feature pipeline: numeric branch + embedded state branch -> Xbf[S][B][512]
// ---------------------------------------------------------------------------
__global__ void __launch_bounds__(128) feature_kernel(
    const float* __restrict__ x_num, const int* __restrict__ x_state,
    const float* __restrict__ emb,
    const unsigned short* wn0, const float* bn0,
    const unsigned short* wn1, const float* bn1,
    const unsigned short* ws0, const float* bs0,
    const unsigned short* ws1, const float* bs1,
    unsigned short* __restrict__ xbf) {
  __shared__ unsigned short lds[4][16 * LSTR];
  int wave = threadIdx.x >> 5, lane = threadIdx.x & 31;
  int tok0 = (blockIdx.x * 4 + wave) * 16;
  int mrow = tok0 + (lane & 15);

  v16bf a0[2];
  // numeric branch A fragments straight from f32 input
  {
    const float* xp = x_num + (size_t)mrow * 64;
    #pragma unroll
    for (int kc = 0; kc < 2; ++kc) {
      int kb = kc * 32 + ((lane >> 4) << 3);
      union { unsigned short s[16]; v16bf b; } u;
      #pragma unroll
      for (int e = 0; e < 8; ++e) {
        u.s[e]     = f32_to_bf16(xp[kb + e]);
        u.s[e + 8] = f32_to_bf16(xp[kb + 16 + e]);
      }
      a0[kc] = u.b;
    }
  }
  mlp_branch(a0, wn0, bn0, wn1, bn1, lds[wave], xbf, tok0, 0, lane);

  // state branch: embedding gather -> A fragments (K index = slot*4 + dim)
  {
    const int* sp = x_state + (size_t)mrow * 16;
    #pragma unroll
    for (int kc = 0; kc < 2; ++kc) {
      int kb = kc * 32 + ((lane >> 4) << 3);
      union { unsigned short s[16]; v16bf b; } u;
      #pragma unroll
      for (int e = 0; e < 8; ++e) {
        int k1 = kb + e, k2 = kb + 16 + e;
        u.s[e]     = f32_to_bf16(emb[sp[k1 >> 2] * 4 + (k1 & 3)]);
        u.s[e + 8] = f32_to_bf16(emb[sp[k2 >> 2] * 4 + (k2 & 3)]);
      }
      a0[kc] = u.b;
    }
  }
  mlp_branch(a0, ws0, bs0, ws1, bs1, lds[wave], xbf, tok0, FEAT, lane);
}

// ---------------------------------------------------------------------------
// Persistent LSTM: 256 waves, one 16x16 h-tile each; c-state in VGPRs.
// Per step: [128,1024(x_t|h)] x [1024 -> 2048(4 gates)] via 128 WMMAs/wave,
// then gate fusion, bf16 h write (double buffered), grid barrier.
// ---------------------------------------------------------------------------
__device__ __forceinline__ void grid_sync(unsigned* counter, unsigned target) {
  __syncthreads();
  if (threadIdx.x == 0) {
    __hip_atomic_fetch_add(counter, 1u, __ATOMIC_ACQ_REL, __HIP_MEMORY_SCOPE_AGENT);
    while (__hip_atomic_load(counter, __ATOMIC_ACQUIRE, __HIP_MEMORY_SCOPE_AGENT) < target)
      __builtin_amdgcn_s_sleep(4);
  }
  __syncthreads();
}

__device__ __forceinline__ float sigmoidf_(float x) {
  return 1.f / (1.f + __expf(-x));
}

__global__ void __launch_bounds__(128) lstm_kernel(
    const unsigned short* __restrict__ xbf,   // [S][128][512] bf16
    const unsigned short* __restrict__ wcat,  // [2048][1024] bf16 (Wx|Wh)
    const float* __restrict__ biasg,          // [2048]
    unsigned short* __restrict__ hbuf,        // [2][128][512] bf16
    unsigned* counter) {
  int wave = threadIdx.x >> 5, lane = threadIdx.x & 31;
  int gw = blockIdx.x * 4 + wave;             // 0..255
  int m0 = (gw >> 5) << 4;                    // batch tile
  int n0 = (gw & 31) << 4;                    // hidden tile
  int nlane = n0 + (lane & 15);

  float bi_ = biasg[nlane];
  float bf_ = biasg[512 + nlane];
  float bg_ = biasg[1024 + nlane];
  float bo_ = biasg[1536 + nlane];

  v8f cacc = {};                              // cell state, persistent
  for (int t = 0; t < S_DIM; ++t) {
    const unsigned short* hin  = hbuf + (size_t)(t & 1) * (B_DIM * HID);
    unsigned short*       hout = hbuf + (size_t)((t + 1) & 1) * (B_DIM * HID);
    const unsigned short* xt   = xbf + (size_t)t * (B_DIM * XDIM);

    // pull next step's x rows toward L2/L0 while this step computes
    if (t + 1 < S_DIM) {
      const unsigned short* xn = xt + (size_t)(B_DIM * XDIM) +
                                 (size_t)(m0 + (lane & 15)) * XDIM +
                                 ((lane >> 4) << 8);
      __builtin_prefetch(xn, 0, 1);
    }

    v8f gi, gf, gg, go;
    #pragma unroll
    for (int r = 0; r < 8; ++r) { gi[r] = bi_; gf[r] = bf_; gg[r] = bg_; go[r] = bo_; }

    // x_t part (K = 0..511 of Wcat rows): group A + all 4 B loads per chunk
    #pragma unroll
    for (int kc = 0; kc < 16; ++kc) {
      v16bf a  = load_a_frag(xt, XDIM, m0, kc * 32, lane);
      v16bf b0 = load_b_frag(wcat, 1024, n0,        kc * 32, lane);
      v16bf b1 = load_b_frag(wcat, 1024, 512 + n0,  kc * 32, lane);
      v16bf b2 = load_b_frag(wcat, 1024, 1024 + n0, kc * 32, lane);
      v16bf b3 = load_b_frag(wcat, 1024, 1536 + n0, kc * 32, lane);
      gi = wmma_bf16(gi, a, b0);
      gf = wmma_bf16(gf, a, b1);
      gg = wmma_bf16(gg, a, b2);
      go = wmma_bf16(go, a, b3);
    }
    // h part (K = 512..1023 of Wcat rows)
    #pragma unroll
    for (int kc = 0; kc < 16; ++kc) {
      v16bf a  = load_a_frag(hin, HID, m0, kc * 32, lane);
      v16bf b0 = load_b_frag(wcat, 1024, n0,        512 + kc * 32, lane);
      v16bf b1 = load_b_frag(wcat, 1024, 512 + n0,  512 + kc * 32, lane);
      v16bf b2 = load_b_frag(wcat, 1024, 1024 + n0, 512 + kc * 32, lane);
      v16bf b3 = load_b_frag(wcat, 1024, 1536 + n0, 512 + kc * 32, lane);
      gi = wmma_bf16(gi, a, b0);
      gf = wmma_bf16(gf, a, b1);
      gg = wmma_bf16(gg, a, b2);
      go = wmma_bf16(go, a, b3);
    }
    // gate fusion + h write
    #pragma unroll
    for (int r = 0; r < 8; ++r) {
      float iv = sigmoidf_(gi[r]);
      float fv = sigmoidf_(gf[r]);
      float gv = tanhf(gg[r]);
      float ov = sigmoidf_(go[r]);
      float c  = fv * cacc[r] + iv * gv;
      cacc[r]  = c;
      float h  = ov * tanhf(c);
      int M = m0 + ((lane >> 4) << 3) + r;
      hout[(size_t)M * HID + nlane] = f32_to_bf16(h);
    }
    __threadfence();
    grid_sync(counter, (unsigned)(t + 1) * NBLK);
  }
}

// ---------------------------------------------------------------------------
// Output projection: h_last[128,512] @ out_w.T[512,32] + out_b -> f32 out
// ---------------------------------------------------------------------------
__global__ void __launch_bounds__(512) out_kernel(
    const unsigned short* __restrict__ hbf,   // buffer 0 holds h_last
    const unsigned short* __restrict__ wout,  // [32,512] bf16
    const float* __restrict__ outb, float* __restrict__ out) {
  int wave = threadIdx.x >> 5, lane = threadIdx.x & 31;
  int m0 = (wave >> 1) << 4, n0 = (wave & 1) << 4;
  v8f acc = {};
  #pragma unroll
  for (int kc = 0; kc < 16; ++kc) {
    v16bf a = load_a_frag(hbf, HID, m0, kc * 32, lane);
    v16bf b = load_b_frag(wout, HID, n0, kc * 32, lane);
    acc = wmma_bf16(acc, a, b);
  }
  float bn = outb[n0 + (lane & 15)];
  #pragma unroll
  for (int r = 0; r < 8; ++r) {
    int M = m0 + ((lane >> 4) << 3) + r;
    out[M * 32 + n0 + (lane & 15)] = acc[r] + bn;
  }
}

// ---------------------------------------------------------------------------
// Prep kernels
// ---------------------------------------------------------------------------
__global__ void init_kernel(unsigned short* hbuf, unsigned* counter) {
  int i = blockIdx.x * blockDim.x + threadIdx.x;
  if (i == 0) *counter = 0u;
  if (i < 2 * B_DIM * HID) hbuf[i] = 0;
}

__global__ void cvt_bf16(const float* __restrict__ src,
                         unsigned short* __restrict__ dst, int n) {
  int i = blockIdx.x * blockDim.x + threadIdx.x;
  if (i < n) dst[i] = f32_to_bf16(src[i]);
}

__global__ void pack_gates(
    const float* Wii, const float* Whi, const float* Wif, const float* Whf,
    const float* Wig, const float* Whg, const float* Wio, const float* Who,
    const float* Bii, const float* Bhi, const float* Bif, const float* Bhf,
    const float* Big, const float* Bhg, const float* Bio, const float* Bho,
    unsigned short* __restrict__ wcat, float* __restrict__ biasg) {
  const float* wx[4] = {Wii, Wif, Wig, Wio};
  const float* wh[4] = {Whi, Whf, Whg, Who};
  int idx = blockIdx.x * blockDim.x + threadIdx.x;
  for (int i = idx; i < 2048 * 1024; i += gridDim.x * blockDim.x) {
    int row = i >> 10, k = i & 1023;
    int g = row >> 9, n = row & 511;
    float v = (k < 512) ? wx[g][n * 512 + k] : wh[g][n * 512 + (k - 512)];
    wcat[i] = f32_to_bf16(v);
  }
  if (idx < 2048) {
    const float* bx[4] = {Bii, Bif, Big, Bio};
    const float* bh[4] = {Bhi, Bhf, Bhg, Bho};
    int g = idx >> 9, n = idx & 511;
    biasg[idx] = bx[g][n] + bh[g][n];
  }
}

// ---------------------------------------------------------------------------
extern "C" void kernel_launch(void* const* d_in, const int* in_sizes, int n_in,
                              void* d_out, int out_size, void* d_ws, size_t ws_size,
                              hipStream_t stream) {
  (void)in_sizes; (void)n_in; (void)out_size; (void)ws_size;
  const float* x_num  = (const float*)d_in[0];
  const int*   x_state= (const int*)d_in[1];
  const float* num_w0 = (const float*)d_in[2];
  const float* num_b0 = (const float*)d_in[3];
  const float* num_w1 = (const float*)d_in[4];
  const float* num_b1 = (const float*)d_in[5];
  const float* emb    = (const float*)d_in[6];
  const float* st_w0  = (const float*)d_in[7];
  const float* st_b0  = (const float*)d_in[8];
  const float* st_w1  = (const float*)d_in[9];
  const float* st_b1  = (const float*)d_in[10];
  const float* out_w  = (const float*)d_in[27];
  const float* out_b  = (const float*)d_in[28];

  char* ws = (char*)d_ws;
  size_t off = 0;
  auto take = [&](size_t bytes) -> void* {
    void* p = ws + off;
    off += (bytes + 255) & ~(size_t)255;
    return p;
  };
  unsigned short* xbf   = (unsigned short*)take((size_t)S_DIM * B_DIM * XDIM * 2);
  unsigned short* hbuf  = (unsigned short*)take((size_t)2 * B_DIM * HID * 2);
  unsigned short* wcat  = (unsigned short*)take((size_t)2048 * 1024 * 2);
  float*          biasg = (float*)take(2048 * 4);
  unsigned short* wn0   = (unsigned short*)take(256 * 64 * 2);
  unsigned short* wn1   = (unsigned short*)take(256 * 256 * 2);
  unsigned short* wst0  = (unsigned short*)take(256 * 64 * 2);
  unsigned short* wst1  = (unsigned short*)take(256 * 256 * 2);
  unsigned short* wou   = (unsigned short*)take(32 * 512 * 2);
  unsigned*       cnt   = (unsigned*)take(256);

  init_kernel<<<(2 * B_DIM * HID + 255) / 256, 256, 0, stream>>>(hbuf, cnt);
  cvt_bf16<<<(256 * 64 + 255) / 256, 256, 0, stream>>>(num_w0, wn0, 256 * 64);
  cvt_bf16<<<(256 * 256 + 255) / 256, 256, 0, stream>>>(num_w1, wn1, 256 * 256);
  cvt_bf16<<<(256 * 64 + 255) / 256, 256, 0, stream>>>(st_w0, wst0, 256 * 64);
  cvt_bf16<<<(256 * 256 + 255) / 256, 256, 0, stream>>>(st_w1, wst1, 256 * 256);
  cvt_bf16<<<(32 * 512 + 255) / 256, 256, 0, stream>>>(out_w, wou, 32 * 512);
  pack_gates<<<2048, 256, 0, stream>>>(
      (const float*)d_in[11], (const float*)d_in[12], (const float*)d_in[13],
      (const float*)d_in[14], (const float*)d_in[15], (const float*)d_in[16],
      (const float*)d_in[17], (const float*)d_in[18],
      (const float*)d_in[19], (const float*)d_in[20], (const float*)d_in[21],
      (const float*)d_in[22], (const float*)d_in[23], (const float*)d_in[24],
      (const float*)d_in[25], (const float*)d_in[26],
      wcat, biasg);

  // 131072 tokens / 16 per wave / 4 waves per block = 2048 blocks
  feature_kernel<<<2048, 128, 0, stream>>>(
      x_num, x_state, emb, wn0, num_b0, wn1, num_b1, wst0, st_b0, wst1, st_b1, xbf);

  lstm_kernel<<<NBLK, 128, 0, stream>>>(xbf, wcat, biasg, hbuf, cnt);

  // after 1024 steps, h_last sits in buffer (1024 & 1) == 0
  out_kernel<<<1, 512, 0, stream>>>(hbuf, wou, out_b, (float*)d_out);
}
